// recommendation_17660905521750
// MI455X (gfx1250) — matrix operand order repaired and checked by reading, not compiled, not run on hardware
//
#include <hip/hip_runtime.h>
#include <hip/hip_bf16.h>

#define SENTINEL 24635

typedef __attribute__((ext_vector_type(16))) _Float16 v16h;
typedef __attribute__((ext_vector_type(8)))  float    v8f;
typedef __attribute__((ext_vector_type(4)))  unsigned u32x4;
typedef __attribute__((ext_vector_type(8)))  unsigned u32x8;

union Frag { v16h v; unsigned u[8]; };

__device__ inline v8f wmma_f16(v16h a, v16h b, v8f c) {
  return __builtin_amdgcn_wmma_f32_16x16x32_f16(
      false, a, false, b, (short)0, c, false, false);
}

// Per-lane async copy of 16B global -> LDS (ASYNCcnt tracked).
__device__ inline void async_copy16(const _Float16* lds, const _Float16* g) {
  unsigned ldsOff = (unsigned)(unsigned long long)(const void*)lds;
  unsigned long long ga = (unsigned long long)(const void*)g;
  asm volatile("global_load_async_to_lds_b128 %0, %1, off"
               :: "v"(ldsOff), "v"(ga) : "memory");
}

// TDM: DMA a 2D tile (64 rows x 128 f16) from a row-major [*,128] tensor into
// LDS with 8-half row padding (128 halves data + 8 pad = USTR stride).
// D# layout per CDNA5 ISA. Issue from ONE wave only.
__device__ inline void tdm_load_user_tile(const _Float16* gsrc,
                                          const _Float16* ldsDst,
                                          unsigned tensorRows) {
  unsigned long long ga = (unsigned long long)(const void*)gsrc;
  u32x4 g0;
  g0[0] = 1u;                                     // count=1, no gather
  g0[1] = (unsigned)(unsigned long long)(const void*)ldsDst;  // lds_addr
  g0[2] = (unsigned)ga;                           // global_addr[31:0]
  g0[3] = (unsigned)(ga >> 32) | (2u << 30);      // global_addr[56:32] | type=2
  u32x8 g1;
  // data_size=2B (code 1), pad_enable, pad_interval=64 DWORDs (code 5),
  // pad_amount=4 DWORDs (code 3) -> 128-half rows padded to 136 halves.
  g1[0] = (1u << 16) | (1u << 20) | (5u << 22) | (3u << 25);
  g1[1] = 128u << 16;                             // tensor_dim0 = 128 (lo16)
  g1[2] = (tensorRows & 0xffffu) << 16;           // tensor_dim1 lo16
  g1[3] = 128u << 16;                             // tile_dim0 = 128
  g1[4] = 64u;                                    // tile_dim1 = 64, tile_dim2=0
  g1[5] = 128u;                                   // tensor_dim0_stride = 128
  g1[6] = 0u;
  g1[7] = 0u;
  asm volatile("tensor_load_to_lds %0, %1" :: "s"(g0), "s"(g1) : "memory");
}

// A fragment (16x32 f16, M x K), contiguous k order: k = 16*half + s.
__device__ inline v16h load_frag_a(const _Float16* lds, int rowBase, int kBase,
                                   int stride, int lane) {
  int half = lane >> 4, m = lane & 15;
  const _Float16* p = lds + (rowBase + m) * stride + kBase;
  Frag f;
#pragma unroll
  for (int j = 0; j < 8; ++j) {
    int k = (half << 4) + (j << 1);
    f.u[j] = *(const unsigned*)(p + k);
  }
  return f.v;
}

// B fragment (32x16 f16, K x N), value(k,n) stored at lds[n*stride + k].
__device__ inline v16h load_frag_b(const _Float16* lds, int nBase, int kBase,
                                   int stride, int lane) {
  int half = lane >> 4, n = lane & 15;
  const _Float16* p = lds + (nBase + n) * stride + kBase + (half << 4);
  Frag f;
#pragma unroll
  for (int j = 0; j < 8; ++j) f.u[j] = *(const unsigned*)(p + (j << 1));
  return f.v;
}

// B fragment with value(k,n) stored at lds[k*stride + n].
__device__ inline v16h load_frag_bt(const _Float16* lds, int nBase, int kBase,
                                    int stride, int lane) {
  int half = lane >> 4, n = lane & 15;
  const _Float16* p = lds + (kBase + (half << 4)) * stride + nBase + n;
  Frag f;
#pragma unroll
  for (int j = 0; j < 8; ++j) {
    unsigned lo = *(const unsigned short*)(p + ((j << 1) + 0) * stride);
    unsigned hi = *(const unsigned short*)(p + ((j << 1) + 1) * stride);
    f.u[j] = lo | (hi << 16);
  }
  return f.v;
}

// ---------------- pooling: ragged gather + masked mean -> f16 user embs -------
__global__ __launch_bounds__(128) void pool_kernel(
    const float* __restrict__ ent_embs, const int* __restrict__ ent_list,
    _Float16* __restrict__ user_h, int L) {
  int b = blockIdx.x, d = threadIdx.x;
  float sum = 0.f; int cnt = 0;
  for (int l = 0; l < L; ++l) {
    int idx = ent_list[b * L + l];
    if (idx == SENTINEL) break;
    sum += ent_embs[(long)idx * 128 + d];
    ++cnt;
  }
  float v = (cnt == 0) ? ent_embs[(long)SENTINEL * 128 + d] : sum / (float)cnt;
  user_h[(long)b * 128 + d] = (_Float16)v;
}

// ---------------- f32 -> f16 convert ------------------------------------------
__global__ void f32_to_f16_kernel(const float* __restrict__ in,
                                  _Float16* __restrict__ out, long n) {
  long i = blockIdx.x * (long)blockDim.x + threadIdx.x;
  long stride = (long)gridDim.x * blockDim.x;
  for (; i < n; i += stride) out[i] = (_Float16)in[i];
}

// ---------------- GEMM: out[M,N] = act(A[M,K] @ W[N,K]^T) * scale -------------
// Double-buffered LDS tiles filled by per-lane async global->LDS DMA.
#define GTS 40  // padded LDS row stride in halves
__global__ __launch_bounds__(256)
__attribute__((amdgpu_waves_per_eu(2))) void gemm_wmma_kernel(
    const _Float16* __restrict__ A, const _Float16* __restrict__ W,
    _Float16* __restrict__ out, int M, int N, int K, int doRelu, float outScale) {
  __shared__ _Float16 sA[2][128 * GTS];
  __shared__ _Float16 sB[2][128 * GTS];
  int tid = threadIdx.x, lane = tid & 31, wave = tid >> 5;
  int rowBlk = blockIdx.y * 128, colBlk = blockIdx.x * 128;
  v8f zero = {0.f, 0.f, 0.f, 0.f, 0.f, 0.f, 0.f, 0.f};
  v8f acc[8];
#pragma unroll
  for (int n = 0; n < 8; ++n) acc[n] = zero;

  auto issue = [&](int buf, int kt) {
    const _Float16* gA = A + (long)rowBlk * K + kt * 32;
    const _Float16* gW = W + (long)colBlk * K + kt * 32;
#pragma unroll
    for (int j = 0; j < 2; ++j) {          // 512 16B chunks / 256 threads
      int c = tid + (j << 8);
      int r = c >> 2, ko = (c & 3) << 3;
      async_copy16(sA[buf] + r * GTS + ko, gA + (long)r * K + ko);
      async_copy16(sB[buf] + r * GTS + ko, gW + (long)r * K + ko);
    }
  };

  int nk = K / 32;
  issue(0, 0);
  for (int kt = 0; kt < nk; ++kt) {
    asm volatile("s_wait_asynccnt 0x0" ::: "memory");  // my chunks of tile kt landed
    __syncthreads();  // everyone's landed; all done computing tile kt-1
    if (kt + 1 < nk) issue((kt + 1) & 1, kt + 1);      // overlaps with compute below
    const _Float16* cA = sA[kt & 1];
    const _Float16* cB = sB[kt & 1];
    // All fragment loads first, then a scheduling fence, then the WMMA burst.
    // The fence keeps the loads from being sunk next to their consumers, so all
    // fragments get distinct registers and the waits become partial.
    v16h aF = load_frag_a(cA, wave * 16, 0, GTS, lane);
    v16h bF[8];
#pragma unroll
    for (int n = 0; n < 8; ++n) bF[n] = load_frag_b(cB, n * 16, 0, GTS, lane);
    __builtin_amdgcn_sched_barrier(0);
#pragma unroll
    for (int n = 0; n < 8; ++n) acc[n] = wmma_f16(aF, bF[n], acc[n]);
  }

  int half = lane >> 4, m = lane & 15;
  int row0 = rowBlk + wave * 16 + half * 8;
#pragma unroll
  for (int n = 0; n < 8; ++n) {
    int col = colBlk + n * 16 + m;
#pragma unroll
    for (int i = 0; i < 8; ++i) {
      float v = acc[n][i] * outScale;
      if (doRelu) v = v > 0.f ? v : 0.f;
      out[(long)(row0 + i) * N + col] = (_Float16)v;
    }
  }
}

// ---------------- fused flash attention: softmax(q @ u^T) @ u -----------------
// Key/value tiles DMA'd by the Tensor Data Mover, double buffered.
#define USTR 136   // user tile LDS stride (halves): 128 data + 8 TDM pad
#define PSTR 72    // P tile LDS stride (halves)
__global__ __launch_bounds__(256)
__attribute__((amdgpu_waves_per_eu(2))) void attn_kernel(
    const _Float16* __restrict__ q, const _Float16* __restrict__ user,
    float* __restrict__ out, int B) {
  __shared__ _Float16 sU[2][64 * USTR];
  __shared__ _Float16 sP[128 * PSTR];
  int tid = threadIdx.x, lane = tid & 31, wave = tid >> 5;
  int half = lane >> 4, m = lane & 15;
  int rowBlk = blockIdx.x * 128;

  // Q fragments in registers: 16 rows/wave, K=128 -> 4 fragments.
  v16h aQ[4];
  {
    const _Float16* qp = q + (long)(rowBlk + wave * 16 + m) * 128;
#pragma unroll
    for (int ks = 0; ks < 4; ++ks) {
      Frag f;
#pragma unroll
      for (int j = 0; j < 8; ++j) {
        int k = ks * 32 + (half << 4) + (j << 1);
        f.u[j] = *(const unsigned*)(qp + k);
      }
      aQ[ks] = f.v;
    }
  }

  v8f zero = {0.f, 0.f, 0.f, 0.f, 0.f, 0.f, 0.f, 0.f};
  float mrow[8], lrow[8];
  v8f o[8];
#pragma unroll
  for (int i = 0; i < 8; ++i) { mrow[i] = -3.4e38f; lrow[i] = 0.f; o[i] = zero; }

  int nTiles = B / 64;
  if (wave == 0) tdm_load_user_tile(user, &sU[0][0], (unsigned)B);

  for (int kt = 0; kt < nTiles; ++kt) {
    if (wave == 0) __builtin_amdgcn_s_wait_tensorcnt(0);  // tile kt DMA done
    __syncthreads();  // tile kt visible; all waves done with tile kt-1's buffer
    if (wave == 0 && kt + 1 < nTiles)
      tdm_load_user_tile(user + (long)(kt + 1) * 64 * 128,
                         &sU[(kt + 1) & 1][0], (unsigned)B);
    const _Float16* cU = sU[kt & 1];

    // S = Q @ U^T: per k-step, load 4 B frags, fence, 4-WMMA burst.
    v8f s[4];
#pragma unroll
    for (int n = 0; n < 4; ++n) s[n] = zero;
#pragma unroll
    for (int ks = 0; ks < 4; ++ks) {
      v16h bF[4];
#pragma unroll
      for (int n = 0; n < 4; ++n) bF[n] = load_frag_b(cU, n * 16, ks * 32, USTR, lane);
      __builtin_amdgcn_sched_barrier(0);
#pragma unroll
      for (int n = 0; n < 4; ++n) s[n] = wmma_f16(aQ[ks], bF[n], s[n]);
    }

    // online softmax row stats + P spill (wave-private rows of sP)
#pragma unroll
    for (int i = 0; i < 8; ++i) {
      float v = s[0][i];
#pragma unroll
      for (int n = 1; n < 4; ++n) v = fmaxf(v, s[n][i]);
#pragma unroll
      for (int off = 1; off < 16; off <<= 1) v = fmaxf(v, __shfl_xor(v, off, 32));
      float mn = fmaxf(mrow[i], v);
      float alpha = __expf(mrow[i] - mn);
      mrow[i] = mn;
      lrow[i] *= alpha;
#pragma unroll
      for (int n = 0; n < 8; ++n) o[n][i] *= alpha;
      float ls = 0.f;
      int prow = wave * 16 + i + half * 8;
#pragma unroll
      for (int n = 0; n < 4; ++n) {
        float p = __expf(s[n][i] - mn);
        ls += p;
        sP[prow * PSTR + n * 16 + m] = (_Float16)p;
      }
#pragma unroll
      for (int off = 1; off < 16; off <<= 1) ls += __shfl_xor(ls, off, 32);
      lrow[i] += ls;
    }
    asm volatile("s_wait_dscnt 0x0" ::: "memory");  // wave-local P visibility

    // O += P @ U: per k-step, load aP + 8 B frags, fence, 8-WMMA burst.
#pragma unroll
    for (int ks = 0; ks < 2; ++ks) {
      v16h aP = load_frag_a(sP, wave * 16, ks * 32, PSTR, lane);
      v16h bF[8];
#pragma unroll
      for (int n = 0; n < 8; ++n) bF[n] = load_frag_bt(cU, n * 16, ks * 32, USTR, lane);
      __builtin_amdgcn_sched_barrier(0);
#pragma unroll
      for (int n = 0; n < 8; ++n) o[n] = wmma_f16(aP, bF[n], o[n]);
    }
  }

  int row0 = rowBlk + wave * 16 + half * 8;
#pragma unroll
  for (int n = 0; n < 8; ++n) {
    int col = n * 16 + m;
#pragma unroll
    for (int i = 0; i < 8; ++i)
      out[(long)(row0 + i) * 128 + col] = o[n][i] / lrow[i];
  }
}

extern "C" void kernel_launch(void* const* d_in, const int* in_sizes, int n_in,
                              void* d_out, int out_size, void* d_ws, size_t ws_size,
                              hipStream_t stream) {
  (void)in_sizes; (void)n_in; (void)out_size; (void)ws_size;
  const float* enc      = (const float*)d_in[0];   // [4096,1024]
  const float* ent_embs = (const float*)d_in[1];   // [24636,128]
  const float* W1       = (const float*)d_in[2];   // [768,1024]
  const float* W2       = (const float*)d_in[3];   // [512,768]
  const float* W3       = (const float*)d_in[4];   // [128,512]
  const int*   ent_list = (const int*)d_in[5];     // [4096,50]
  float* out = (float*)d_out;                      // [4096,128]

  const int B = 4096, L = 50;
  char* ws = (char*)d_ws;
  size_t off = 0;
  auto alloc = [&](size_t bytes) -> void* {
    void* p = ws + off; off += (bytes + 255) & ~(size_t)255; return p;
  };
  _Float16* enc_h  = (_Float16*)alloc((size_t)B * 1024 * 2);
  _Float16* w1_h   = (_Float16*)alloc((size_t)768 * 1024 * 2);
  _Float16* w2_h   = (_Float16*)alloc((size_t)512 * 768 * 2);
  _Float16* w3_h   = (_Float16*)alloc((size_t)128 * 512 * 2);
  _Float16* x1_h   = (_Float16*)alloc((size_t)B * 768 * 2);
  _Float16* x2_h   = (_Float16*)alloc((size_t)B * 512 * 2);
  _Float16* q_h    = (_Float16*)alloc((size_t)B * 128 * 2);
  _Float16* user_h = (_Float16*)alloc((size_t)B * 128 * 2);

  pool_kernel<<<dim3(B), dim3(128), 0, stream>>>(ent_embs, ent_list, user_h, L);

  f32_to_f16_kernel<<<dim3(1024), dim3(256), 0, stream>>>(enc, enc_h, (long)B * 1024);
  f32_to_f16_kernel<<<dim3(512),  dim3(256), 0, stream>>>(W1, w1_h, 768L * 1024);
  f32_to_f16_kernel<<<dim3(512),  dim3(256), 0, stream>>>(W2, w2_h, 512L * 768);
  f32_to_f16_kernel<<<dim3(256),  dim3(256), 0, stream>>>(W3, w3_h, 128L * 512);

  dim3 blk(256);
  gemm_wmma_kernel<<<dim3(768 / 128, B / 128), blk, 0, stream>>>(
      enc_h, w1_h, x1_h, B, 768, 1024, 1, 1.0f);
  gemm_wmma_kernel<<<dim3(512 / 128, B / 128), blk, 0, stream>>>(
      x1_h, w2_h, x2_h, B, 512, 768, 1, 1.0f);
  gemm_wmma_kernel<<<dim3(128 / 128, B / 128), blk, 0, stream>>>(
      x2_h, w3_h, q_h, B, 128, 512, 0, 0.08838834764831845f);

  attn_kernel<<<dim3(B / 128), blk, 0, stream>>>(q_h, user_h, out, B);
}
